// Tokenizer_54709293416603
// MI455X (gfx1250) — matrix operand order repaired and checked by reading, not compiled, not run on hardware
//
#include <hip/hip_runtime.h>

typedef _Float16 v16h __attribute__((ext_vector_type(16)));
typedef _Float16 v8h  __attribute__((ext_vector_type(8)));
typedef float    v8f  __attribute__((ext_vector_type(8)));

#define BATCH 16
#define V_DIM 64
#define S_DIM 8192
#define S_PAD 8448          // pad so t-tiles up to t=511 read in-bounds (max s = 8239)
#define D_DIM 128           // SAMPLE_DIM
#define P_DIM 512           // PATCH_DIM
#define WIN   64
#define STEP  16
#define T_DIM 508           // len(arange(0, 8192-64, 16))
#define K_DIM 8192          // D_DIM * WIN, == W_patch row length
#define NKS   256           // K_DIM / 32 k-slabs

// ---------------------------------------------------------------------------
// Stage 1: ns[b,d,s] = floor(x[b,:,s] . W_sample[d,:] + b_sample[d]) -> f16
// ---------------------------------------------------------------------------
__global__ __launch_bounds__(256) void sample_lsh_kernel(
    const float* __restrict__ x, const float* __restrict__ Ws,
    const float* __restrict__ bs, _Float16* __restrict__ ns)
{
    __shared__ float lds_w[D_DIM * V_DIM];   // 32 KB
    __shared__ float lds_x[V_DIM * 256];     // 64 KB
    __shared__ float lds_b[D_DIM];

    const int tid = threadIdx.x;
    const int b  = blockIdx.y;
    const int s  = blockIdx.x * 256 + tid;
    const bool valid = (s < S_DIM);

    for (int i = tid; i < D_DIM * V_DIM; i += 256) lds_w[i] = Ws[i];
    if (tid < D_DIM) lds_b[tid] = bs[tid];
    for (int v = 0; v < V_DIM; ++v)
        lds_x[v * 256 + tid] = valid ? x[((size_t)b * V_DIM + v) * S_DIM + s] : 0.0f;
    __syncthreads();

    for (int d = 0; d < D_DIM; ++d) {
        float acc = lds_b[d];
        #pragma unroll
        for (int v = 0; v < V_DIM; ++v)
            acc = fmaf(lds_x[v * 256 + tid], lds_w[d * V_DIM + v], acc);
        float val = valid ? floorf(acc) : 0.0f;   // integer-valued, exact in f16
        ns[((size_t)b * D_DIM + d) * S_PAD + s] = (_Float16)val;
    }
}

// ---------------------------------------------------------------------------
// Stage 2a: split W_patch into f16 hi + lo (hi+lo ~= f32 exact)
// ---------------------------------------------------------------------------
__global__ __launch_bounds__(256) void wsplit_kernel(
    const float* __restrict__ Wp,
    _Float16* __restrict__ whi, _Float16* __restrict__ wlo)
{
    size_t i = (size_t)blockIdx.x * 256 + threadIdx.x;
    float w = Wp[i];
    _Float16 hi = (_Float16)w;
    whi[i] = hi;
    wlo[i] = (_Float16)(w - (float)hi);
}

// ---------------------------------------------------------------------------
// Stage 2b: tokens[b,t,p] = floor( A(t x 8192k) @ B(8192k x p) + b_patch )
//   A[t,k] = ns[b, k/64, t*16 + k%64]  (f16, exact);  k-slab ks covers
//   k = ks*32 .. ks*32+31  (d = ks>>1, w0 = (ks&1)*32).
//   B[k,p] = W_patch[p,k] = whi + wlo  (split f16 -> 2 WMMAs per slab)
// Block: 256 thr = 8 waves, arranged 2 (t) x 4 (p); each wave computes a
// 2x2 grid of 16x16 C tiles (32t x 32p) -> block tile 64t x 128p.
// Fragments are explicitly double-buffered (prefetch ks+1 while computing ks)
// so v_wmma does not serialize on s_wait_loadcnt 0x0.
// ---------------------------------------------------------------------------
struct Frags { v16h a0, a1, bh0, bh1, bl0, bl1; };

__device__ __forceinline__ v16h ld_fragA(const _Float16* p) {
    // A 16x32 frag in this code's contiguous k order (same as B): lane holds 16 consecutive k (s-offset),
    // already shifted by half*16.
    v8h lo = *(const v8h*)p;
    v8h hi = *(const v8h*)(p + 8);
    return __builtin_shufflevector(lo, hi,
        0,1,2,3,4,5,6,7,8,9,10,11,12,13,14,15);
}
__device__ __forceinline__ v16h ld_fragB(const _Float16* p) {
    // B 32x16 frag in the same contiguous k order as A: lane holds 16 consecutive k at its p-column.
    v8h lo = *(const v8h*)p;
    v8h hi = *(const v8h*)(p + 8);
    return __builtin_shufflevector(lo, hi,
        0,1,2,3,4,5,6,7,8,9,10,11,12,13,14,15);
}
__device__ __forceinline__ v8f wmma_f16(v16h a, v16h b, v8f c) {
    return __builtin_amdgcn_wmma_f32_16x16x32_f16(
        false, a, false, b, (short)0, c, false, false);
}

__global__ __launch_bounds__(256) void patch_gemm_kernel(
    const _Float16* __restrict__ ns, const _Float16* __restrict__ whi,
    const _Float16* __restrict__ wlo, const float* __restrict__ bp,
    float* __restrict__ out)
{
    const int tid  = threadIdx.x;
    const int lane = tid & 31;
    const int wave = tid >> 5;
    const int col  = lane & 15;      // M-row for A frag, N-col for B/C frags
    const int half = lane >> 4;

    const int b  = blockIdx.z;
    const int t0 = blockIdx.x * 64  + (wave & 1) * 32;   // wave: t-tiles t0, t0+16
    const int p0 = blockIdx.y * 128 + (wave >> 1) * 32;  // wave: p-tiles p0, p0+16

    v8f c00 = {}, c01 = {}, c10 = {}, c11 = {};

    const _Float16* nsb = ns + (size_t)b * D_DIM * S_PAD;
    const int sA = (t0 + col) * STEP + half * 16;

    const _Float16* bhiP0 = whi + (size_t)(p0 + col) * K_DIM + half * 16;
    const _Float16* bhiP1 = whi + (size_t)(p0 + 16 + col) * K_DIM + half * 16;
    const _Float16* bloP0 = wlo + (size_t)(p0 + col) * K_DIM + half * 16;
    const _Float16* bloP1 = wlo + (size_t)(p0 + 16 + col) * K_DIM + half * 16;

    auto loadf = [&](int ks) {
        Frags f;
        const _Float16* aP = nsb + (size_t)(ks >> 1) * S_PAD + (ks & 1) * 32 + sA;
        f.a0 = ld_fragA(aP);          // t-tile 0
        f.a1 = ld_fragA(aP + 256);    // t-tile 1 (t0+16 -> +256 in s)
        const int kb = ks * 32;
        f.bh0 = ld_fragB(bhiP0 + kb);
        f.bh1 = ld_fragB(bhiP1 + kb);
        f.bl0 = ld_fragB(bloP0 + kb);
        f.bl1 = ld_fragB(bloP1 + kb);
        return f;
    };

    Frags cur = loadf(0);
    #pragma unroll 2
    for (int ks = 0; ks < NKS; ++ks) {
        Frags nxt = loadf(ks < NKS - 1 ? ks + 1 : NKS - 1);  // branchless prefetch
        c00 = wmma_f16(cur.a0, cur.bh0, c00);
        c00 = wmma_f16(cur.a0, cur.bl0, c00);
        c01 = wmma_f16(cur.a0, cur.bh1, c01);
        c01 = wmma_f16(cur.a0, cur.bl1, c01);
        c10 = wmma_f16(cur.a1, cur.bh0, c10);
        c10 = wmma_f16(cur.a1, cur.bl0, c10);
        c11 = wmma_f16(cur.a1, cur.bh1, c11);
        c11 = wmma_f16(cur.a1, cur.bl1, c11);
        cur = nxt;
    }

    // Epilogue: C tile layout: VGPR r -> M = r + 8*half, N = col.
    v8f* cs[4] = { &c00, &c01, &c10, &c11 };
    #pragma unroll
    for (int i = 0; i < 2; ++i) {
        #pragma unroll
        for (int j = 0; j < 2; ++j) {
            const int p = p0 + j * 16 + col;
            const float bpv = bp[p];
            const v8f& cc = *cs[i * 2 + j];
            #pragma unroll
            for (int r = 0; r < 8; ++r) {
                const int t = t0 + i * 16 + r + half * 8;
                if (t < T_DIM)
                    out[((size_t)b * T_DIM + t) * P_DIM + p] = floorf(cc[r] + bpv);
            }
        }
    }
}

// ---------------------------------------------------------------------------
// Stage 3: LayerNorm over p=512, in-place on tokens. One block per (b,t) row.
// ---------------------------------------------------------------------------
__global__ __launch_bounds__(256) void layernorm_kernel(
    float* __restrict__ out, const float* __restrict__ gamma,
    const float* __restrict__ beta)
{
    __shared__ float ssum[256];
    __shared__ float ssq[256];
    const int tid = threadIdx.x;
    float* rowp = out + (size_t)blockIdx.x * P_DIM;
    float v0 = rowp[tid];
    float v1 = rowp[tid + 256];
    ssum[tid] = v0 + v1;
    ssq[tid]  = v0 * v0 + v1 * v1;
    __syncthreads();
    for (int st = 128; st > 0; st >>= 1) {
        if (tid < st) { ssum[tid] += ssum[tid + st]; ssq[tid] += ssq[tid + st]; }
        __syncthreads();
    }
    const float mu  = ssum[0] * (1.0f / P_DIM);
    const float var = ssq[0] * (1.0f / P_DIM) - mu * mu;
    const float inv = 1.0f / sqrtf(var + 1e-5f);
    rowp[tid]       = (v0 - mu) * inv * gamma[tid]       + beta[tid];
    rowp[tid + 256] = (v1 - mu) * inv * gamma[tid + 256] + beta[tid + 256];
}

// ---------------------------------------------------------------------------
extern "C" void kernel_launch(void* const* d_in, const int* in_sizes, int n_in,
                              void* d_out, int out_size, void* d_ws, size_t ws_size,
                              hipStream_t stream)
{
    (void)in_sizes; (void)n_in; (void)out_size; (void)ws_size;
    const float* x     = (const float*)d_in[0];
    const float* Ws    = (const float*)d_in[1];
    const float* bs    = (const float*)d_in[2];
    const float* Wp    = (const float*)d_in[3];
    const float* bp    = (const float*)d_in[4];
    const float* gamma = (const float*)d_in[5];
    const float* beta  = (const float*)d_in[6];
    float* out = (float*)d_out;

    _Float16* ns  = (_Float16*)d_ws;                         // 16*128*8448 f16
    _Float16* whi = ns  + (size_t)BATCH * D_DIM * S_PAD;     // 512*8192 f16
    _Float16* wlo = whi + (size_t)P_DIM * K_DIM;             // 512*8192 f16

    sample_lsh_kernel<<<dim3(S_PAD / 256, BATCH), 256, 0, stream>>>(x, Ws, bs, ns);
    wsplit_kernel<<<(P_DIM * K_DIM) / 256, 256, 0, stream>>>(Wp, whi, wlo);
    patch_gemm_kernel<<<dim3(8, 4, BATCH), 256, 0, stream>>>(ns, whi, wlo, bp, out);
    layernorm_kernel<<<BATCH * T_DIM, 256, 0, stream>>>(out, gamma, beta);
}
